// HybridSparseAttention_61864708931957
// MI455X (gfx1250) — matrix operand order repaired and checked by reading, not compiled, not run on hardware
//
#include <hip/hip_runtime.h>
#include <hip/hip_bf16.h>
#include <math.h>

// ---------------------------------------------------------------------------
// HybridSparseAttention for MI455X (gfx1250, wave32, WMMA + async-to-LDS)
//
//   1) f32 -> f16 conversion of x and the 4 weight matrices (workspace)
//   2) Q/K/V projections: WMMA f32_16x16x32_f16 GEMMs (x @ W^T), with the
//      shared A slab staged into LDS via GLOBAL_LOAD_ASYNC_TO_LDS_B128
//      (double-buffered, s_wait_asynccnt) and 16x64 C per wave.
//   3) build gidx/gvalid from global_mask
//   4) attention: block per (b,h,t); 65 local + 32 global scores, two
//      softmaxes (ctx path keeps zero-padded keys; full_attn path masks
//      j >= win_len with -inf, scatters at cols left+j), full_attn row in LDS
//   5) out = ctx @ Wo^T via the same WMMA GEMM into d_out[0 .. B*T*D)
// ---------------------------------------------------------------------------

#define B_  2
#define T_  1024
#define D_  1024
#define H_  16
#define DH_ 64
#define WIN_ 32
#define KSZ_ 65
#define MG_ 32
#define NS_ (KSZ_ + MG_)   // 97 scores per query

typedef __attribute__((ext_vector_type(16))) _Float16 v16h;
typedef __attribute__((ext_vector_type(8)))  _Float16 v8h;
typedef __attribute__((ext_vector_type(8)))  float    v8f;
typedef int v4i __attribute__((ext_vector_type(4)));

// address-space-qualified int4 typedefs (attribute survives on a typedef)
typedef __attribute__((address_space(1))) v4i as1_v4i;   // global
typedef __attribute__((address_space(3))) v4i as3_v4i;   // LDS

// ---------------------------------------------------------------------------
// async 16-byte global -> LDS copy (per lane) with synchronous fallback
// ---------------------------------------------------------------------------
__device__ __forceinline__ void async_b128(const _Float16* g, _Float16* l) {
#if defined(__has_builtin) && __has_builtin(__builtin_amdgcn_global_load_async_to_lds_b128)
    __builtin_amdgcn_global_load_async_to_lds_b128(
        (as1_v4i*)g, (as3_v4i*)l, 0, 0);
#else
    *(v8h*)l = *(const v8h*)g;   // fallback: plain load + ds store
#endif
}

// ---------------------------------------------------------------------------
// f32 -> f16 conversion
// ---------------------------------------------------------------------------
__global__ void cvt_f32_to_f16(const float* __restrict__ in,
                               _Float16* __restrict__ out, int n) {
    int i = blockIdx.x * blockDim.x + threadIdx.x;
    if (i < n) out[i] = (_Float16)in[i];
}

// ---------------------------------------------------------------------------
// C[M,N] = A[M,K] @ Wt[N,K]^T.
// Block = 128 threads (4 waves); block tile 16(M) x 256(N); wave tile 16x64
// (4 accumulators sharing one A fragment). A slab (16x32 halves = 1KB/step)
// staged in LDS via async copy, double buffered. grid = (M/16, N/256).
// ---------------------------------------------------------------------------
__global__ __launch_bounds__(128)
void gemm_f16_wmma(const _Float16* __restrict__ A,
                   const _Float16* __restrict__ Wt,
                   float* __restrict__ C,
                   int M, int N, int K) {
    __shared__ _Float16 Asl[2][16 * 32];   // ping-pong A slab

    const int tid  = threadIdx.x;
    const int lane = tid & 31;
    const int wave = tid >> 5;
    const int m0   = blockIdx.x * 16;
    const int n0   = blockIdx.y * 256 + wave * 64;
    const int l15  = lane & 15;
    const int hsel = lane >> 4;          // 0: lanes 0-15, 1: lanes 16-31
    const int nsteps = K / 32;

    // staging assignment: thread i (0..63): row i>>2, 8-half chunk i&3
    const int arow = tid >> 2;
    const int acol = (tid & 3) * 8;
    const _Float16* gsrc = A + (size_t)(m0 + arow) * K + acol;
    _Float16* lb0 = &Asl[0][arow * 32 + acol];
    _Float16* lb1 = &Asl[1][arow * 32 + acol];

    if (tid < 64) async_b128(gsrc, lb0);          // preload slab 0

    const _Float16* bP = Wt + (size_t)(n0 + l15) * K + 8 * hsel;
    v8f acc[4] = {v8f{}, v8f{}, v8f{}, v8f{}};

    for (int s = 0; s < nsteps; ++s) {
        const int kk = s * 32;
        const _Float16* asrc = &Asl[s & 1][l15 * 32 + 8 * hsel];

        __syncthreads();                           // prev compute done -> buffer reusable
        if (s + 1 < nsteps) {
            if (tid < 64)
                async_b128(gsrc + (kk + 32), (s & 1) ? lb0 : lb1);
            asm volatile("s_wait_asynccnt 0x1" ::: "memory");  // slab s resident
        } else {
            asm volatile("s_wait_asynccnt 0x0" ::: "memory");
        }
        __syncthreads();                           // publish slab s to all waves

        // A fragment from LDS: elems 0..7 = A[m][kk+8h..], 8..15 = A[m][kk+16+8h..]
        v16h a;
        *((v8h*)&a)     = *(const v8h*)(asrc);
        *((v8h*)&a + 1) = *(const v8h*)(asrc + 16);

        const _Float16* bk = bP + kk;
        if (kk + 32 < K) __builtin_prefetch(bk + 32, 0, 3);
#pragma unroll
        for (int sub = 0; sub < 4; ++sub) {
            // B fragment: elems 0..7 = Wt[n][kk+8h..], 8..15 = Wt[n][kk+16+8h..]
            const _Float16* bs = bk + (size_t)(sub * 16) * K;
            v16h b;
            *((v8h*)&b)     = *(const v8h*)(bs);
            *((v8h*)&b + 1) = *(const v8h*)(bs + 16);
            acc[sub] = __builtin_amdgcn_wmma_f32_16x16x32_f16(
                           false, a, false, b, (short)0, acc[sub], false, false);
        }
    }

    // C/D layout: VGPR r -> row m0 + 8*hsel + r, col base + (lane&15)
    float* crow = C + (size_t)(m0 + 8 * hsel) * N + (n0 + l15);
#pragma unroll
    for (int sub = 0; sub < 4; ++sub)
#pragma unroll
        for (int r = 0; r < 8; ++r)
            crow[(size_t)r * N + sub * 16] = acc[sub][r];
}

// ---------------------------------------------------------------------------
// gidx/gvalid: stable argsort(!mask) -> true indices first (in index order),
// then false indices; first MG_ kept. numpy bool -> 1 byte per element.
// ---------------------------------------------------------------------------
__global__ void build_gidx(const unsigned char* __restrict__ mask,
                           int* __restrict__ gidx, float* __restrict__ gvalid) {
    if (threadIdx.x != 0) return;
    int b = blockIdx.x;
    const unsigned char* m = mask + (size_t)b * T_;
    int cnt = 0;
    for (int t = 0; t < T_ && cnt < MG_; ++t)
        if (m[t]) { gidx[b * MG_ + cnt] = t; gvalid[b * MG_ + cnt] = 1.0f; ++cnt; }
    for (int t = 0; t < T_ && cnt < MG_; ++t)
        if (!m[t]) { gidx[b * MG_ + cnt] = t; gvalid[b * MG_ + cnt] = 0.0f; ++cnt; }
}

// ---------------------------------------------------------------------------
// Attention: one block (128 threads) per (b, h, t).
// ---------------------------------------------------------------------------
__global__ __launch_bounds__(128)
void attn_kernel(const float* __restrict__ Q,
                 const float* __restrict__ Kmat,
                 const float* __restrict__ Vmat,
                 const int* __restrict__ gidx,
                 const float* __restrict__ gvalid,
                 float* __restrict__ ctx,
                 float* __restrict__ full_attn) {
    const int t = blockIdx.x, h = blockIdx.y, b = blockIdx.z;
    const int tid = threadIdx.x;

    __shared__ float q[DH_];
    __shared__ float s0[NS_];     // ctx-path scores (padded keys: score 0)
    __shared__ float s1[NS_];     // full_attn-path scores (-inf beyond win_len)
    __shared__ float w0[NS_];
    __shared__ float w1[NS_];
    __shared__ float row[T_];
    __shared__ int   gI[MG_];
    __shared__ float gV[MG_];

    const size_t qoff = ((size_t)(b * T_ + t)) * D_ + h * DH_;
    if (tid < DH_) q[tid] = Q[qoff + tid];
    if (tid < MG_) { gI[tid] = gidx[b * MG_ + tid]; gV[tid] = gvalid[b * MG_ + tid]; }
    for (int i = tid; i < T_; i += blockDim.x) row[i] = 0.0f;
    __syncthreads();

    const int left    = (t - WIN_) > 0 ? (t - WIN_) : 0;
    const int right   = (t + WIN_ + 1) < T_ ? (t + WIN_ + 1) : T_;
    const int win_len = right - left;
    const float scale = 0.125f;   // dh^-0.5

    if (tid < KSZ_) {
        int tp = t - WIN_ + tid;
        float sc = 0.0f;                       // zero-padded key -> score 0
        if (tp >= 0 && tp < T_) {
            const float* kr = Kmat + ((size_t)(b * T_ + tp)) * D_ + h * DH_;
            float acc = 0.0f;
#pragma unroll
            for (int i = 0; i < DH_; ++i) acc += q[i] * kr[i];
            sc = acc * scale;
        }
        s0[tid] = sc;
        s1[tid] = (tid < win_len) ? sc : -__builtin_inff();
    } else if (tid < NS_) {
        int g  = tid - KSZ_;
        int ti = gI[g];
        float gv = gV[g];
        const float* kr = Kmat + ((size_t)(b * T_ + ti)) * D_ + h * DH_;
        float acc = 0.0f;
#pragma unroll
        for (int i = 0; i < DH_; ++i) acc += q[i] * kr[i];
        float sc = acc * scale * gv;                 // Kg *= gvalid before dot
        float sm = (gv > 0.0f) ? sc : -1.0e9f;       // where(gvalid, s, -1e9)
        s0[tid] = sm;
        s1[tid] = sm;
    }
    __syncthreads();

    // two serial softmaxes over 97 scores (threads 0 and 1 in parallel)
    if (tid < 2) {
        const float* sp = tid ? s1 : s0;
        float* wp = tid ? w1 : w0;
        float mx = -__builtin_inff();
        for (int i = 0; i < NS_; ++i) mx = sp[i] > mx ? sp[i] : mx;
        float sum = 0.0f;
        for (int i = 0; i < NS_; ++i) { float e = expf(sp[i] - mx); wp[i] = e; sum += e; }
        float inv = 1.0f / sum;
        for (int i = 0; i < NS_; ++i) wp[i] *= inv;
    }
    __syncthreads();

    // ctx[d] = sum_j w0[j]*Vw[j][d] + sum_g w0[KSZ+g]*gv*Vg[g][d]
    if (tid < DH_) {
        float acc = 0.0f;
        for (int j = 0; j < KSZ_; ++j) {
            int tp = t - WIN_ + j;
            if (tp >= 0 && tp < T_)
                acc += w0[j] * Vmat[((size_t)(b * T_ + tp)) * D_ + h * DH_ + tid];
        }
        for (int g = 0; g < MG_; ++g)
            acc += w0[KSZ_ + g] * gV[g] *
                   Vmat[((size_t)(b * T_ + gI[g])) * D_ + h * DH_ + tid];
        ctx[qoff + tid] = acc;
    }

    // full_attn row: wl_d at cols left+j (distinct cols), then wg_d at gidx
    if (tid < win_len) row[left + tid] += w1[tid];
    __syncthreads();
    if (tid < MG_ && gV[tid] > 0.0f) row[gI[tid]] += w1[KSZ_ + tid];
    __syncthreads();

    float* orow = full_attn + (((size_t)b * H_ + h) * T_ + t) * (size_t)T_;
    for (int i = tid; i < T_; i += blockDim.x) orow[i] = row[i];
}

// ---------------------------------------------------------------------------
// Launch
// ---------------------------------------------------------------------------
extern "C" void kernel_launch(void* const* d_in, const int* in_sizes, int n_in,
                              void* d_out, int out_size, void* d_ws, size_t ws_size,
                              hipStream_t stream) {
    (void)in_sizes; (void)n_in; (void)out_size; (void)ws_size;

    const float* x    = (const float*)d_in[0];
    const unsigned char* gmask = (const unsigned char*)d_in[1];  // bool array
    const float* Wq   = (const float*)d_in[2];
    const float* Wk   = (const float*)d_in[3];
    const float* Wv   = (const float*)d_in[4];
    const float* Wo   = (const float*)d_in[5];
    // d_in[6] = maxG (==32, compile-time constant here)

    const size_t M  = (size_t)B_ * T_;        // 2048
    const size_t NX = M * D_;                 // x / Q / K / V / ctx elems
    const size_t NW = (size_t)D_ * D_;        // weight elems

    char* ws = (char*)d_ws;
    _Float16* xh   = (_Float16*)(ws);                        // 4 MB
    _Float16* Wqh  = (_Float16*)(ws + (4u  << 20));          // 2 MB each
    _Float16* Wkh  = (_Float16*)(ws + (6u  << 20));
    _Float16* Wvh  = (_Float16*)(ws + (8u  << 20));
    _Float16* Woh  = (_Float16*)(ws + (10u << 20));
    float*    Qf   = (float*)   (ws + (12u << 20));          // 8 MB each
    float*    Kf   = (float*)   (ws + (20u << 20));
    float*    Vf   = (float*)   (ws + (28u << 20));
    float*    ctxf = (float*)   (ws + (36u << 20));
    _Float16* ctxh = (_Float16*)(ws + (44u << 20));          // 4 MB
    int*      gI   = (int*)     (ws + (48u << 20));
    float*    gV   = (float*)   (ws + (48u << 20) + 4096);

    float* outp  = (float*)d_out;                // (B,T,D)  = 2M floats
    float* fullp = outp + NX;                    // (B,H,T,T) = 32M floats

    // 1) conversions
    cvt_f32_to_f16<<<(int)((NX + 255) / 256), 256, 0, stream>>>(x,  xh,  (int)NX);
    cvt_f32_to_f16<<<(int)((NW + 255) / 256), 256, 0, stream>>>(Wq, Wqh, (int)NW);
    cvt_f32_to_f16<<<(int)((NW + 255) / 256), 256, 0, stream>>>(Wk, Wkh, (int)NW);
    cvt_f32_to_f16<<<(int)((NW + 255) / 256), 256, 0, stream>>>(Wv, Wvh, (int)NW);
    cvt_f32_to_f16<<<(int)((NW + 255) / 256), 256, 0, stream>>>(Wo, Woh, (int)NW);

    // 2) Q/K/V projections (WMMA, async-LDS staged A)
    dim3 ggrid((unsigned)(M / 16), (unsigned)(D_ / 256));
    gemm_f16_wmma<<<ggrid, 128, 0, stream>>>(xh, Wqh, Qf, (int)M, D_, D_);
    gemm_f16_wmma<<<ggrid, 128, 0, stream>>>(xh, Wkh, Kf, (int)M, D_, D_);
    gemm_f16_wmma<<<ggrid, 128, 0, stream>>>(xh, Wvh, Vf, (int)M, D_, D_);

    // 3) global token indices
    build_gidx<<<B_, 32, 0, stream>>>(gmask, gI, gV);

    // 4) attention + full_attn rows
    dim3 agrid(T_, H_, B_);
    attn_kernel<<<agrid, 128, 0, stream>>>(Qf, Kf, Vf, gI, gV, ctxf, fullp);

    // 5) out = ctx @ Wo^T (WMMA)
    cvt_f32_to_f16<<<(int)((NX + 255) / 256), 256, 0, stream>>>(ctxf, ctxh, (int)NX);
    gemm_f16_wmma<<<ggrid, 128, 0, stream>>>(ctxh, Woh, outp, (int)M, D_, D_);
}
